// GCN_25391846654560
// MI455X (gfx1250) — matrix operand order repaired and checked by reading, not compiled, not run on hardware
//
#include <hip/hip_runtime.h>
#include <hip/hip_bf16.h>

#define EPS_MSG 1e-7f
#define EPS_BN  1e-5f

typedef __attribute__((ext_vector_type(16))) _Float16 v16h;
typedef __attribute__((ext_vector_type(8)))  float    v8f;
typedef int v4i_vs __attribute__((vector_size(16)));   // matches builtin param type

#define AS1 __attribute__((address_space(1)))
#define AS3 __attribute__((address_space(3)))

#if defined(__HIP_DEVICE_COMPILE__) && \
    __has_builtin(__builtin_amdgcn_global_load_async_to_lds_b128) && \
    __has_builtin(__builtin_amdgcn_s_wait_asynccnt)
#define USE_ASYNC_LDS 1
#else
#define USE_ASYNC_LDS 0
#endif

// ---------------------------------------------------------------------------
// Weight packer: W[K,CO] f32 row-major -> WMMA B-fragment order, f16.
// Fragment tile (kt, tn) occupies 512 halves; within a tile, lane L element e
// maps to  k = kt*32 + (L>>4)*8 + (e&7) + (e>>3)*16,  col = tn*16 + (L&15)   (ISA 7.12.2).
// k >= K or col >= CO pad with 0 (this also neutralizes A K-tail garbage).
// ---------------------------------------------------------------------------
__global__ void pack_w_kernel(const float* __restrict__ W, _Float16* __restrict__ Wp,
                              int K, int CO)
{
    const int tiles_n = (CO + 15) >> 4;
    const int ksteps  = (K + 31) >> 5;
    const long total  = (long)ksteps * tiles_n * 512;
    long i = blockIdx.x * (long)blockDim.x + threadIdx.x;
    if (i >= total) return;
    int r    = (int)(i & 511);
    int tile = (int)(i >> 9);
    int tn   = tile % tiles_n;
    int kt   = tile / tiles_n;
    int lane = r >> 4;
    int e    = r & 15;
    int k    = (kt << 5) + ((lane >> 4) << 3) + (e & 7) + ((e >> 3) << 4);
    int col  = (tn << 4) + (lane & 15);
    float v  = (k < K && col < CO) ? W[(long)k * CO + col] : 0.0f;
    Wp[i] = (_Float16)v;
}

// ---------------------------------------------------------------------------
// WMMA GEMM:  C[N,CO] = act(A[N,K]) @ W + bias, W pre-packed to fragments.
// Block = one 16-row M tile; wave w owns output column tile w
// (launch with blockDim = 32 * tiles_n, tiles_n = ceil(CO/16) <= 8).
// A tiles are moved global->LDS with the CDNA5 async-copy engine (ASYNCcnt)
// using a 2-deep ring, overlapping the next tile's DMA with this tile's
// fragment convert + WMMA. Requires N % 16 == 0 (true: N = 100000); K-tail
// reads pull finite garbage that multiplies zero-padded W fragments -> 0.
// ---------------------------------------------------------------------------
__global__ void __launch_bounds__(256)
gemm_wmma_f16(const float* __restrict__ A,
              const _Float16* __restrict__ Wp,
              const float* __restrict__ bias,
              float* __restrict__ C,
              int N, int K, int CO,
              const float* __restrict__ a_scale,
              const float* __restrict__ a_shift)
{
    const int tiles_n = (CO + 15) >> 4;
    const int rowbase = blockIdx.x * 16;
    const int wave    = threadIdx.x >> 5;           // == output column tile
    const int lane    = threadIdx.x & 31;
    const int ml      = lane & 15;
    const int hi      = lane >> 4;
    const int ksteps  = (K + 31) >> 5;
    const v16h* wp    = (const v16h*)Wp;
    v8f acc = {};

#if USE_ASYNC_LDS
    // Two raw 16x32 f32 tiles (row-major), filled by async DMA.
    __shared__ __align__(16) float lds_raw[2][512];

    // One tile = 128 x 16B chunks; chunk i: row = i>>3, 4-float segment i&7.
    // Every K here is a multiple of 4 floats, so all chunks are 16B aligned.
    auto issue_tile = [&](int kt, int buf) {
        const long base = (long)rowbase * K + (long)kt * 32;
        for (int i = threadIdx.x; i < 128; i += blockDim.x) {
            int row = i >> 3, seg = i & 7;
            const float* g = A + base + (long)row * K + seg * 4;
            __builtin_amdgcn_global_load_async_to_lds_b128(
                (AS1 v4i_vs*)g, (AS3 v4i_vs*)&lds_raw[buf][i * 4], 0, 0);
        }
    };

    issue_tile(0, 0);
    for (int kt = 0; kt < ksteps; ++kt) {
        __builtin_amdgcn_s_wait_asynccnt(0);   // this thread's tile-kt chunks done
        __syncthreads();                       // whole tile visible to all waves
        if (kt + 1 < ksteps) issue_tile(kt + 1, (kt + 1) & 1);  // overlap DMA

        // Fragment assembly: lane ml/hi reads two contiguous 8-float runs
        // (kk = hi*8 + e and kk = 16 + hi*8 + e) -> ds_load_b128 pairs.
        const float* rowp = &lds_raw[kt & 1][ml * 32 + hi * 8];
        const int k0 = kt << 5;
        v16h a;
#pragma unroll
        for (int e = 0; e < 8; ++e) {
            float v0 = rowp[e];
            float v1 = rowp[16 + e];
            if (a_scale) {                     // folded BatchNorm + ReLU
                int ka0 = k0 + hi * 8 + e;
                int ka1 = ka0 + 16;
                v0 = v0 * a_scale[ka0] + a_shift[ka0]; v0 = v0 > 0.0f ? v0 : 0.0f;
                v1 = v1 * a_scale[ka1] + a_shift[ka1]; v1 = v1 > 0.0f ? v1 : 0.0f;
            }
            a[e]     = (_Float16)v0;
            a[8 + e] = (_Float16)v1;
        }
        v16h b = wp[(size_t)(kt * tiles_n + wave) * 32 + lane];
        acc = __builtin_amdgcn_wmma_f32_16x16x32_f16(false, a, false, b,
                                                     (short)0, acc, false, false);
    }
#else
    // Fallback: synchronous LDS staging in fragment order (round-2 path).
    __shared__ __align__(32) _Float16 lds_a[512];
    for (int kt = 0; kt < ksteps; ++kt) {
        const int k0 = kt << 5;
        __syncthreads();
        for (int idx = threadIdx.x; idx < 512; idx += blockDim.x) {
            int m   = idx >> 5;
            int kk  = idx & 31;
            int row = rowbase + m;
            int ka  = k0 + kk;
            float av = 0.0f;
            if (row < N && ka < K) {
                av = A[(long)row * K + ka];
                if (a_scale) {
                    av = av * a_scale[ka] + a_shift[ka];
                    av = av > 0.0f ? av : 0.0f;
                }
            }
            int h2 = (kk >> 3) & 1;
            int kp = kk - (h2 << 3);
            int e  = (kp < 8) ? kp : kp - 8;
            lds_a[((m + (h2 << 4)) << 4) + e] = (_Float16)av;
        }
        __syncthreads();
        v16h a = ((const v16h*)lds_a)[lane];
        v16h b = wp[(size_t)(kt * tiles_n + wave) * 32 + lane];
        acc = __builtin_amdgcn_wmma_f32_16x16x32_f16(false, a, false, b,
                                                     (short)0, acc, false, false);
    }
#endif

    const int bcol = (wave << 4) + ml;
    if (bcol < CO) {
        const float bv = bias ? bias[bcol] : 0.0f;
#pragma unroll
        for (int r = 0; r < 8; ++r) {
            // C/D layout: lanes 0-15 hold M=r, lanes 16-31 hold M=r+8.
            int orow = rowbase + r + (hi << 3);
            if (orow < N) C[(long)orow * CO + bcol] = acc[r] + bv;
        }
    }
}

// ---------------------------------------------------------------------------
__global__ void fill_f32(float* __restrict__ p, float v, long n)
{
    long i = blockIdx.x * (long)blockDim.x + threadIdx.x;
    if (i < n) p[i] = v;
}

// Pass A: per-(dst, feature) max of logits. Messages are relu(h)+eps > 0, so
// float ordering == uint-bit ordering and atomicMax on uint bits is exact.
__global__ void edge_max_kernel(const float* __restrict__ h,
                                const int* __restrict__ src,
                                const int* __restrict__ dst,
                                unsigned* __restrict__ mx,
                                const float* __restrict__ t, int E, int C)
{
    long i = blockIdx.x * (long)blockDim.x + threadIdx.x;
    if (i >= (long)E * C) return;
    int e = (int)(i / C);
    int c = (int)(i - (long)e * C);
    const float* hp = &h[(long)src[e] * C];
    __builtin_prefetch(hp, 0, 1);                    // global_prefetch_b8
    float v = hp[c];
    float msg = (v > 0.0f ? v : 0.0f) + EPS_MSG;
    float logit = t[0] * msg;
    atomicMax(&mx[(long)dst[e] * C + c], __float_as_uint(logit));
}

// Pass B: ex = exp(logit - mx[dst]); denom += ex; num += ex*msg.
__global__ void edge_sum_kernel(const float* __restrict__ h,
                                const int* __restrict__ src,
                                const int* __restrict__ dst,
                                const unsigned* __restrict__ mx,
                                float* __restrict__ denom,
                                float* __restrict__ num,
                                const float* __restrict__ t, int E, int C)
{
    long i = blockIdx.x * (long)blockDim.x + threadIdx.x;
    if (i >= (long)E * C) return;
    int e = (int)(i / C);
    int c = (int)(i - (long)e * C);
    const float* hp = &h[(long)src[e] * C];
    __builtin_prefetch(hp, 0, 1);
    float v = hp[c];
    float msg = (v > 0.0f ? v : 0.0f) + EPS_MSG;
    float logit = t[0] * msg;
    long  di = (long)dst[e] * C + c;
    float ex = __expf(logit - __uint_as_float(mx[di]));
    atomicAdd(&denom[di], ex);
    atomicAdd(&num[di], ex * msg);
}

// In place: num = num/max(denom,1e-16) + proj   (softmax aggr + residual h)
__global__ void combine_kernel(float* __restrict__ num,
                               const float* __restrict__ denom,
                               const float* __restrict__ proj, long n)
{
    long i = blockIdx.x * (long)blockDim.x + threadIdx.x;
    if (i < n) num[i] = num[i] / fmaxf(denom[i], 1e-16f) + proj[i];
}

// Training-mode BatchNorm stats per column, folded into scale/shift.
__global__ void bn_stats_kernel(const float* __restrict__ z, int N, int C2,
                                const float* __restrict__ gamma,
                                const float* __restrict__ beta,
                                float* __restrict__ scale,
                                float* __restrict__ shift)
{
    __shared__ float s1[256], s2[256];
    const int c = blockIdx.x;
    float sum = 0.0f, sq = 0.0f;
    for (int i = threadIdx.x; i < N; i += blockDim.x) {
        float v = z[(long)i * C2 + c];
        sum += v; sq += v * v;
    }
    s1[threadIdx.x] = sum; s2[threadIdx.x] = sq;
    __syncthreads();
    for (int s = 128; s > 0; s >>= 1) {
        if ((int)threadIdx.x < s) {
            s1[threadIdx.x] += s1[threadIdx.x + s];
            s2[threadIdx.x] += s2[threadIdx.x + s];
        }
        __syncthreads();
    }
    if (threadIdx.x == 0) {
        float mu  = s1[0] / (float)N;
        float var = s2[0] / (float)N - mu * mu;
        float sc  = gamma[c] * rsqrtf(var + EPS_BN);
        scale[c] = sc;
        shift[c] = beta[c] - mu * sc;
    }
}

// hout[n, :] = concat(mlp[n, :co], x[n, :4]) * mask[n, :]
__global__ void concat_dropout_kernel(const float* __restrict__ mlp, int co,
                                      const float* __restrict__ x,
                                      const float* __restrict__ mask,
                                      float* __restrict__ out, int N)
{
    const int Wd = co + 4;
    long i = blockIdx.x * (long)blockDim.x + threadIdx.x;
    if (i >= (long)N * Wd) return;
    int n = (int)(i / Wd);
    int j = (int)(i - (long)n * Wd);
    float v = (j < co) ? mlp[(long)n * co + j] : x[(long)n * 4 + (j - co)];
    out[i] = v * mask[i];
}

__global__ void deg_kernel(const int* __restrict__ dst, float* __restrict__ deg, int E)
{
    int e = blockIdx.x * blockDim.x + threadIdx.x;
    if (e < E) atomicAdd(&deg[dst[e]], 1.0f);
}

__global__ void sage_gather_kernel(const float* __restrict__ h,
                                   const int* __restrict__ src,
                                   const int* __restrict__ dst,
                                   float* __restrict__ snb, int E)
{
    long i = blockIdx.x * (long)blockDim.x + threadIdx.x;
    if (i >= (long)E * 12) return;
    int e = (int)(i / 12);
    int d = (int)(i - (long)e * 12);
    atomicAdd(&snb[(long)dst[e] * 12 + d], h[(long)src[e] * 12 + d]);
}

__global__ void sage_out_kernel(const float* __restrict__ snb,
                                const float* __restrict__ deg,
                                const float* __restrict__ h,
                                const float* __restrict__ wl,
                                const float* __restrict__ bl,
                                const float* __restrict__ wr,
                                float* __restrict__ out, int N)
{
    int n = blockIdx.x * blockDim.x + threadIdx.x;
    if (n >= N) return;
    float dg  = fmaxf(deg[n], 1.0f);
    float acc = bl[0];
#pragma unroll
    for (int j = 0; j < 12; ++j)
        acc += (snb[(long)n * 12 + j] / dg) * wl[j] + h[(long)n * 12 + j] * wr[j];
    out[n] = acc;
}

// ---------------------------------------------------------------------------
static inline size_t pack_halves(int K, int CO)
{
    return (size_t)((K + 31) / 32) * ((CO + 15) / 16) * 512;
}

extern "C" void kernel_launch(void* const* d_in, const int* in_sizes, int n_in,
                              void* d_out, int out_size, void* d_ws, size_t ws_size,
                              hipStream_t stream)
{
    (void)in_sizes; (void)n_in; (void)out_size; (void)ws_size;
    const int N = 100000, E = 1600000;
    static const int CI[7] = {4, 12, 20, 36, 68, 36, 20};
    static const int CO[7] = {8, 16, 32, 64, 32, 16, 8};

    // Assumed dict-insertion-order flattening of setup_inputs():
    //   x, edge_index, 7 x (lin_w, lin_b, t, w1, b1, gamma, beta, w2, b2),
    //   sage_wl, sage_bl, sage_wr, 7 dropout masks.
    const float* x   = (const float*)d_in[0];
    const int*   ei  = (const int*)d_in[1];
    const int*   src = ei;
    const int*   dst = ei + E;
    const float *lw[7], *lb[7], *lt[7], *w1[7], *b1[7], *gm[7], *bt[7], *w2[7], *b2[7];
    int p = 2;
    for (int l = 0; l < 7; ++l) {
        lw[l] = (const float*)d_in[p++]; lb[l] = (const float*)d_in[p++];
        lt[l] = (const float*)d_in[p++];
        w1[l] = (const float*)d_in[p++]; b1[l] = (const float*)d_in[p++];
        gm[l] = (const float*)d_in[p++]; bt[l] = (const float*)d_in[p++];
        w2[l] = (const float*)d_in[p++]; b2[l] = (const float*)d_in[p++];
    }
    const float* sage_wl = (const float*)d_in[p++];
    const float* sage_bl = (const float*)d_in[p++];
    const float* sage_wr = (const float*)d_in[p++];
    const float* mask[7];
    for (int l = 0; l < 7; ++l) mask[l] = (const float*)d_in[p++];

    // ---- workspace carve-up ----
    // Packed f16 weight fragments first (1KB-granular, keeps 32B alignment).
    _Float16* hws = (_Float16*)d_ws;
    _Float16 *plin[7], *pw1[7], *pw2[7];
    size_t hoff = 0;
    for (int l = 0; l < 7; ++l) {
        const int ci = CI[l], co = CO[l], c2 = 2 * CO[l];
        plin[l] = hws + hoff; hoff += pack_halves(ci, co);
        pw1[l]  = hws + hoff; hoff += pack_halves(co, c2);
        pw2[l]  = hws + hoff; hoff += pack_halves(c2, co);
    }
    // f32 buffers. All per-node arrays fit in the 192MB L2 -> edge gathers
    // and segment-reduction atomics below stay cache-resident.
    float* ws    = (float*)(hws + ((hoff + 15) & ~(size_t)15));
    float* hA    = ws; ws += (size_t)N * 72;
    float* hB    = ws; ws += (size_t)N * 72;
    float* proj  = ws; ws += (size_t)N * 64;   // lin_src output; reused as MLP out
    float* mxb   = ws; ws += (size_t)N * 64;   // segment max (float bits in uint)
    float* denom = ws; ws += (size_t)N * 64;
    float* num   = ws; ws += (size_t)N * 64;   // becomes aggr + h (in place)
    float* z     = ws; ws += (size_t)N * 128;
    float* scale = ws; ws += 128;
    float* shift = ws; ws += 128;
    float* deg   = ws; ws += (size_t)N;
    float* snb   = ws; ws += (size_t)N * 12;

    // ---- pack all weights into WMMA fragment order (cheap; runs per call) ----
    for (int l = 0; l < 7; ++l) {
        const int ci = CI[l], co = CO[l], c2 = 2 * CO[l];
        long t0 = (long)pack_halves(ci, co);
        long t1 = (long)pack_halves(co, c2);
        long t2 = (long)pack_halves(c2, co);
        pack_w_kernel<<<(unsigned)((t0 + 255) / 256), 256, 0, stream>>>(lw[l], plin[l], ci, co);
        pack_w_kernel<<<(unsigned)((t1 + 255) / 256), 256, 0, stream>>>(w1[l], pw1[l], co, c2);
        pack_w_kernel<<<(unsigned)((t2 + 255) / 256), 256, 0, stream>>>(w2[l], pw2[l], c2, co);
    }

    const unsigned mblocks = (N + 15) / 16;     // N % 16 == 0 -> exact cover
    const float* hin = x;
    for (int l = 0; l < 7; ++l) {
        const int ci = CI[l], co = CO[l], c2 = 2 * co;
        const unsigned bd_co = 32u * ((co + 15) / 16);
        const unsigned bd_c2 = 32u * ((c2 + 15) / 16);

        // 1) proj = hin @ lin_w + lin_b          (WMMA)
        gemm_wmma_f16<<<mblocks, bd_co, 0, stream>>>(
            hin, plin[l], lb[l], proj, N, ci, co, nullptr, nullptr);

        // 2) softmax aggregation over edges
        const long nc = (long)N * co;
        fill_f32<<<(unsigned)((nc + 255) / 256), 256, 0, stream>>>(mxb,   0.0f, nc);
        fill_f32<<<(unsigned)((nc + 255) / 256), 256, 0, stream>>>(denom, 0.0f, nc);
        fill_f32<<<(unsigned)((nc + 255) / 256), 256, 0, stream>>>(num,   0.0f, nc);
        const long ec = (long)E * co;
        edge_max_kernel<<<(unsigned)((ec + 255) / 256), 256, 0, stream>>>(
            proj, src, dst, (unsigned*)mxb, lt[l], E, co);
        edge_sum_kernel<<<(unsigned)((ec + 255) / 256), 256, 0, stream>>>(
            proj, src, dst, (const unsigned*)mxb, denom, num, lt[l], E, co);
        combine_kernel<<<(unsigned)((nc + 255) / 256), 256, 0, stream>>>(
            num, denom, proj, nc);

        // 3) MLP: z = (aggr+h) @ w1 + b1         (WMMA)
        gemm_wmma_f16<<<mblocks, bd_c2, 0, stream>>>(
            num, pw1[l], b1[l], z, N, co, c2, nullptr, nullptr);
        //    BatchNorm training stats -> folded scale/shift
        bn_stats_kernel<<<c2, 256, 0, stream>>>(z, N, c2, gm[l], bt[l], scale, shift);
        //    mlp_out(proj) = relu(bn(z)) @ w2 + b2  (WMMA, BN+ReLU fused on A)
        gemm_wmma_f16<<<mblocks, bd_co, 0, stream>>>(
            z, pw2[l], b2[l], proj, N, c2, co, scale, shift);

        // 4) concat skip + inverted dropout
        float* ho = (l & 1) ? hB : hA;
        const long nw = (long)N * (co + 4);
        concat_dropout_kernel<<<(unsigned)((nw + 255) / 256), 256, 0, stream>>>(
            proj, co, x, mask[l], ho, N);
        hin = ho;
    }

    // Final SAGEConv(12 -> 1): mean neighbor + root transform
    fill_f32<<<(unsigned)((N + 255) / 256), 256, 0, stream>>>(deg, 0.0f, (long)N);
    fill_f32<<<(unsigned)(((long)N * 12 + 255) / 256), 256, 0, stream>>>(snb, 0.0f, (long)N * 12);
    deg_kernel<<<(unsigned)((E + 255) / 256), 256, 0, stream>>>(dst, deg, E);
    sage_gather_kernel<<<(unsigned)(((long)E * 12 + 255) / 256), 256, 0, stream>>>(
        hin, src, dst, snb, E);
    sage_out_kernel<<<(unsigned)((N + 255) / 256), 256, 0, stream>>>(
        snb, deg, hin, sage_wl, sage_bl, sage_wr, (float*)d_out, N);
}
